// GNNEncoder_84550726189713
// MI455X (gfx1250) — matrix operand order repaired and checked by reading, not compiled, not run on hardware
//
#include <hip/hip_runtime.h>
#include <hip/hip_bf16.h>
#include <stdint.h>

// ---- problem constants (from reference) ----
#define N_NODES 20000
#define N_EDGES 200000
#define EA_EDGES (N_EDGES + N_NODES)   // with self loops
#define DMODEL 256
#define NHEAD 8
#define CHEAD 32
#define NLAYER 3
#define VA 128
#define VB 16
#define NEG_SLOPE 0.2f
#define LN_EPS 1e-5f

typedef _Float16 v16h __attribute__((ext_vector_type(16)));
typedef float    v8f  __attribute__((ext_vector_type(8)));

// ---------- helpers ----------
__device__ __forceinline__ unsigned ordered_u32(float f) {
  unsigned u = __float_as_uint(f);
  return (u & 0x80000000u) ? ~u : (u | 0x80000000u);
}
__device__ __forceinline__ float ordered_f32(unsigned u) {
  return (u & 0x80000000u) ? __uint_as_float(u & 0x7fffffffu)
                           : __uint_as_float(~u);
}

// ---------- embeddings ----------
__global__ void k_atom_embed(const int* __restrict__ x,
                             const float* __restrict__ emb,
                             float* __restrict__ h) {
  int i = blockIdx.x * blockDim.x + threadIdx.x;  // over N*D
  if (i >= N_NODES * DMODEL) return;
  int n = i >> 8, d = i & 255;
  float s = 0.f;
#pragma unroll
  for (int f = 0; f < 9; ++f) {
    int idx = x[n * 9 + f];
    s += emb[(f * VA + idx) * DMODEL + d];
  }
  h[i] = s;
}

__global__ void k_bond_embed(const int* __restrict__ eattr,
                             const float* __restrict__ emb,
                             float* __restrict__ ea) {
  int i = blockIdx.x * blockDim.x + threadIdx.x;  // over E*D
  if (i >= N_EDGES * DMODEL) return;
  int e = i >> 8, d = i & 255;
  float s = 0.f;
#pragma unroll
  for (int f = 0; f < 3; ++f) {
    int idx = eattr[e * 3 + f];
    s += emb[(f * VB + idx) * DMODEL + d];
  }
  ea[i] = s;
}

// ---------- generic utility kernels ----------
__global__ void k_zero_f32(float* __restrict__ p, int count) {
  int i = blockIdx.x * blockDim.x + threadIdx.x;
  if (i < count) p[i] = 0.f;
}
__global__ void k_zero_u32(unsigned* __restrict__ p, int count) {
  int i = blockIdx.x * blockDim.x + threadIdx.x;
  if (i < count) p[i] = 0u;   // ordered encoding of -inf
}
__global__ void k_cvt_f16(const float* __restrict__ src,
                          _Float16* __restrict__ dst, int count) {
  int i = blockIdx.x * blockDim.x + threadIdx.x;
  if (i < count) dst[i] = (_Float16)src[i];
}

// ---------- weight convert + swizzle into WMMA B-fragment order ----------
// Layout (per 256x256 matrix): half index t = ((nT*8 + kb)*32 + lane)*16 + j
//   col = nT*16 + (lane&15);  k = kb*32 + (lane>>4)*16 + j
// so each lane's 16-half B fragment for one WMMA is 32B-contiguous.
__global__ void k_cvt_swizzle_w(const float* __restrict__ W,
                                _Float16* __restrict__ Wsw, int nmat) {
  int i = blockIdx.x * blockDim.x + threadIdx.x;  // over nmat*65536
  if (i >= nmat * 65536) return;
  int mat = i >> 16, t = i & 65535;
  int j    = t & 15;
  int lane = (t >> 4) & 31;
  int kb   = (t >> 9) & 7;
  int nT   = t >> 12;
  int col  = nT * 16 + (lane & 15);
  int k    = kb * 32 + ((lane >> 4) << 4) + j;
  Wsw[i] = (_Float16)W[mat * 65536 + k * 256 + col];
}

// ---------- self-loop mean edge_attr ----------
__global__ void k_edge_count(const int* __restrict__ dst, float* __restrict__ cnt) {
  int e = blockIdx.x * blockDim.x + threadIdx.x;
  if (e < N_EDGES) atomicAdd(&cnt[dst[e]], 1.0f);
}
__global__ void k_loop_sum(const float* __restrict__ ea,
                           const int* __restrict__ dst,
                           float* __restrict__ loop /* = ea + E*D */) {
  int i = blockIdx.x * blockDim.x + threadIdx.x;  // over E*D
  if (i >= N_EDGES * DMODEL) return;
  int e = i >> 8, d = i & 255;
  atomicAdd(&loop[dst[e] * DMODEL + d], ea[i]);
}
__global__ void k_loop_div(float* __restrict__ loop, const float* __restrict__ cnt) {
  int i = blockIdx.x * blockDim.x + threadIdx.x;  // over N*D
  if (i >= N_NODES * DMODEL) return;
  int n = i >> 8;
  float c = cnt[n];
  loop[i] = loop[i] / (c > 1.f ? c : 1.f);
}

// ---------- WMMA GEMM:  C[M x 256] = A16[M x 256] * Wsw (+bias) ----------
// block: 128 threads (4 waves). block tile: 16 rows x 64 cols.
// grid: (M/16, 4).  M multiple of 16 (20000 and 220000 both are).
// A staged in LDS in A-fragment order -> 2 x ds_load_b128 per WMMA.
// B read from pre-swizzled weights    -> 2 x global_load_b128 per WMMA.
__global__ __launch_bounds__(128) void k_gemm_wmma(
    const _Float16* __restrict__ A, const _Float16* __restrict__ Wsw,
    const float* __restrict__ bias, float* __restrict__ Co) {
  // A-fragment order: [kb(8)][laneSlot(32)][j(16)] halves = 8KB
  __shared__ __align__(16) _Float16 As[16 * 256];
  const int tid = threadIdx.x;
  const int rowBase = blockIdx.x * 16;

  // stage A strip (16x256, row-major in global) into fragment order.
  // global uint4 i covers halves [i*8, i*8+8): row m=i/32, chunk q=i%32,
// k0=q*8 -> kb=q>>2, c=q&3; c>>1 -> lane-half, c&1 -> j-quad.
  {
    const uint4* Ag = (const uint4*)(A + (size_t)rowBase * 256);
    uint4* As4 = (uint4*)As;
#pragma unroll
    for (int t = 0; t < 4; ++t) {
      int i = tid + t * 128;
      int m = i >> 5;
      int q = i & 31;
      int kb = q >> 2;
      int c  = q & 3;
      int slot = m + ((c >> 1) << 4);
      As4[kb * 64 + slot * 2 + (c & 1)] = Ag[i];
    }
  }
  __syncthreads();

  const int lane = tid & 31;
  const int wave = tid >> 5;
  const int nTileG = blockIdx.y * 4 + wave;   // 0..15 (16-col tile index)

  v8f acc = {};
  const uint4* As4r = (const uint4*)As;
  const uint4* Bg = (const uint4*)(Wsw + (size_t)nTileG * 4096);

#pragma unroll
  for (int kbi = 0; kbi < 8; ++kbi) {
    union { v16h v; uint4 q[2]; } a, b;
    a.q[0] = As4r[kbi * 64 + lane * 2 + 0];
    a.q[1] = As4r[kbi * 64 + lane * 2 + 1];
    b.q[0] = Bg[(kbi * 32 + lane) * 2 + 0];
    b.q[1] = Bg[(kbi * 32 + lane) * 2 + 1];
    if (kbi < 7) __builtin_prefetch(&Bg[((kbi + 1) * 32 + lane) * 2], 0, 0);
    acc = __builtin_amdgcn_wmma_f32_16x16x32_f16(
        false, a.v, false, b.v, (short)0, acc, false, false);
  }

  const int hi  = lane >> 4;
  const int col = nTileG * 16 + (lane & 15);
  const float bv = bias ? bias[col] : 0.f;
#pragma unroll
  for (int v = 0; v < 8; ++v) {
    int row = rowBase + (hi << 3) + v;   // D layout: vgpr v -> M = v + hi*8
    Co[(size_t)row * DMODEL + col] = acc[v] + bv;
  }
}

// ---------- attention ----------
__global__ void k_score_max(const float* __restrict__ xl, const float* __restrict__ xr,
                            const float* __restrict__ ee, const float* __restrict__ att,
                            const int* __restrict__ esrc, const int* __restrict__ edst,
                            float* __restrict__ score, unsigned* __restrict__ smaxu) {
  int i = blockIdx.x * blockDim.x + threadIdx.x;  // over Ea*H
  if (i >= EA_EDGES * NHEAD) return;
  int e = i >> 3, hh = i & 7;
  int s_, d_;
  if (e < N_EDGES) { s_ = esrc[e]; d_ = edst[e]; }
  else             { s_ = e - N_EDGES; d_ = s_; }
  const float4* pl = (const float4*)(xl + (size_t)s_ * DMODEL + hh * CHEAD);
  const float4* pr = (const float4*)(xr + (size_t)d_ * DMODEL + hh * CHEAD);
  const float4* pe = (const float4*)(ee + (size_t)e * DMODEL + hh * CHEAD);
  const float4* pa = (const float4*)(att + hh * CHEAD);
  float sc = 0.f;
#pragma unroll
  for (int c = 0; c < CHEAD / 4; ++c) {
    float4 L = pl[c], R = pr[c], Ev = pe[c], Av = pa[c];
    float g0 = L.x + R.x + Ev.x; g0 = (g0 > 0.f) ? g0 : NEG_SLOPE * g0;
    float g1 = L.y + R.y + Ev.y; g1 = (g1 > 0.f) ? g1 : NEG_SLOPE * g1;
    float g2 = L.z + R.z + Ev.z; g2 = (g2 > 0.f) ? g2 : NEG_SLOPE * g2;
    float g3 = L.w + R.w + Ev.w; g3 = (g3 > 0.f) ? g3 : NEG_SLOPE * g3;
    sc += g0 * Av.x + g1 * Av.y + g2 * Av.z + g3 * Av.w;
  }
  score[i] = sc;
  atomicMax(&smaxu[d_ * NHEAD + hh], ordered_u32(sc));
}

__global__ void k_exp_sum(float* __restrict__ score, const unsigned* __restrict__ smaxu,
                          float* __restrict__ denom, const int* __restrict__ edst) {
  int i = blockIdx.x * blockDim.x + threadIdx.x;  // over Ea*H
  if (i >= EA_EDGES * NHEAD) return;
  int e = i >> 3, hh = i & 7;
  int d_ = (e < N_EDGES) ? edst[e] : (e - N_EDGES);
  float mx = ordered_f32(smaxu[d_ * NHEAD + hh]);
  float p = __expf(score[i] - mx);
  score[i] = p;
  atomicAdd(&denom[d_ * NHEAD + hh], p);
}

__global__ void k_init_out(float* __restrict__ out, const float* __restrict__ cbias) {
  int i = blockIdx.x * blockDim.x + threadIdx.x;  // over N*D
  if (i >= N_NODES * DMODEL) return;
  out[i] = cbias[i & 255];
}

__global__ void k_aggregate(const float* __restrict__ score, const float* __restrict__ denom,
                            const float* __restrict__ xl,
                            const int* __restrict__ esrc, const int* __restrict__ edst,
                            float* __restrict__ out) {
  int i = blockIdx.x * blockDim.x + threadIdx.x;  // over Ea*H
  if (i >= EA_EDGES * NHEAD) return;
  int e = i >> 3, hh = i & 7;
  int s_, d_;
  if (e < N_EDGES) { s_ = esrc[e]; d_ = edst[e]; }
  else             { s_ = e - N_EDGES; d_ = s_; }
  float alpha = score[i] / denom[d_ * NHEAD + hh];
  const float4* pl = (const float4*)(xl + (size_t)s_ * DMODEL + hh * CHEAD);
  float* po = out + (size_t)d_ * DMODEL + hh * CHEAD;
#pragma unroll
  for (int c = 0; c < CHEAD / 4; ++c) {
    float4 L = pl[c];
    atomicAdd(&po[c * 4 + 0], alpha * L.x);
    atomicAdd(&po[c * 4 + 1], alpha * L.y);
    atomicAdd(&po[c * 4 + 2], alpha * L.z);
    atomicAdd(&po[c * 4 + 3], alpha * L.w);
  }
}

// ---------- LayerNorm + SiLU (one wave32 per node) ----------
__global__ void k_ln_silu(const float* __restrict__ out, const float* __restrict__ g,
                          const float* __restrict__ b, float* __restrict__ h) {
  int node = blockIdx.x * 8 + (threadIdx.x >> 5);
  int lane = threadIdx.x & 31;
  if (node >= N_NODES) return;
  const float4* p4 = (const float4*)(out + (size_t)node * DMODEL + lane * 8);
  float4 u0 = p4[0], u1 = p4[1];
  float v[8] = {u0.x, u0.y, u0.z, u0.w, u1.x, u1.y, u1.z, u1.w};
  float s = 0.f, s2 = 0.f;
#pragma unroll
  for (int i = 0; i < 8; ++i) { s += v[i]; s2 += v[i] * v[i]; }
#pragma unroll
  for (int off = 16; off > 0; off >>= 1) {
    s  += __shfl_xor(s, off, 32);
    s2 += __shfl_xor(s2, off, 32);
  }
  float mu = s * (1.f / DMODEL);
  float var = s2 * (1.f / DMODEL) - mu * mu;
  float inv = rsqrtf(var + LN_EPS);
#pragma unroll
  for (int i = 0; i < 8; ++i) {
    int d = lane * 8 + i;
    float y = (v[i] - mu) * inv * g[d] + b[d];
    h[(size_t)node * DMODEL + d] = y * (1.f / (1.f + __expf(-y)));
  }
}

// ---------- final output (h, then batch as raw int bits) ----------
__global__ void k_finalize(const float* __restrict__ out, const int* __restrict__ batch,
                           float* __restrict__ dout) {
  int i = blockIdx.x * blockDim.x + threadIdx.x;
  const int total = N_NODES * DMODEL + N_NODES;
  if (i >= total) return;
  if (i < N_NODES * DMODEL) dout[i] = out[i];
  else ((int*)dout)[i] = batch[i - N_NODES * DMODEL];
}

// =====================================================================
extern "C" void kernel_launch(void* const* d_in, const int* in_sizes, int n_in,
                              void* d_out, int out_size, void* d_ws, size_t ws_size,
                              hipStream_t stream) {
  const int*   x        = (const int*)  d_in[0];
  const int*   eidx     = (const int*)  d_in[1];
  const int*   eattr    = (const int*)  d_in[2];
  const int*   batch    = (const int*)  d_in[3];
  const float* atom_emb = (const float*)d_in[4];
  const float* bond_emb = (const float*)d_in[5];
  const float* Wl       = (const float*)d_in[6];
  const float* bl       = (const float*)d_in[7];
  const float* Wr       = (const float*)d_in[8];
  const float* br       = (const float*)d_in[9];
  const float* We       = (const float*)d_in[10];
  const float* att      = (const float*)d_in[11];
  const float* cbias    = (const float*)d_in[12];
  const float* ln_g     = (const float*)d_in[13];
  const float* ln_b     = (const float*)d_in[14];
  (void)in_sizes; (void)n_in; (void)out_size; (void)ws_size;

  const int* esrc = eidx;
  const int* edst = eidx + N_EDGES;

  // ---- workspace carve-up (256B aligned) ----
  char* ws = (char*)d_ws;
  size_t off = 0;
  auto carve = [&](size_t bytes) -> char* {
    char* p = ws + off;
    off += (bytes + 255) & ~(size_t)255;
    return p;
  };
  float*     h     = (float*)    carve(sizeof(float)     * N_NODES  * DMODEL);
  _Float16*  h16   = (_Float16*) carve(sizeof(_Float16)  * N_NODES  * DMODEL);
  float*     ea    = (float*)    carve(sizeof(float)     * EA_EDGES * DMODEL);
  _Float16*  ea16  = (_Float16*) carve(sizeof(_Float16)  * EA_EDGES * DMODEL);
  float*     ee    = (float*)    carve(sizeof(float)     * EA_EDGES * DMODEL);
  float*     xl    = (float*)    carve(sizeof(float)     * N_NODES  * DMODEL);
  float*     xr    = (float*)    carve(sizeof(float)     * N_NODES  * DMODEL);
  _Float16*  wsw   = (_Float16*) carve(sizeof(_Float16)  * 3 * NLAYER * DMODEL * DMODEL);
  float*     score = (float*)    carve(sizeof(float)     * EA_EDGES * NHEAD);
  unsigned*  smaxu = (unsigned*) carve(sizeof(unsigned)  * N_NODES  * NHEAD);
  float*     denom = (float*)    carve(sizeof(float)     * N_NODES  * NHEAD);
  float*     outb  = (float*)    carve(sizeof(float)     * N_NODES  * DMODEL);
  float*     cnt   = (float*)    carve(sizeof(float)     * N_NODES);

  _Float16* Wlsw = wsw;
  _Float16* Wrsw = wsw + (size_t)NLAYER * DMODEL * DMODEL;
  _Float16* Wesw = wsw + (size_t)2 * NLAYER * DMODEL * DMODEL;

  const int TB = 256;
  const int ND  = N_NODES  * DMODEL;   // 5,120,000
  const int ED  = N_EDGES  * DMODEL;   // 51,200,000
  const int EAD = EA_EDGES * DMODEL;   // 56,320,000
  const int EAH = EA_EDGES * NHEAD;    // 1,760,000
  const int NH  = N_NODES  * NHEAD;

  // ---- encoders ----
  k_atom_embed<<<ND / TB, TB, 0, stream>>>(x, atom_emb, h);
  k_bond_embed<<<ED / TB, TB, 0, stream>>>(eattr, bond_emb, ea);

  // ---- self-loop mean edge_attr ----
  k_zero_f32<<<(N_NODES + TB - 1) / TB, TB, 0, stream>>>(cnt, N_NODES);
  k_zero_f32<<<ND / TB, TB, 0, stream>>>(ea + (size_t)N_EDGES * DMODEL, ND);
  k_edge_count<<<(N_EDGES + TB - 1) / TB, TB, 0, stream>>>(edst, cnt);
  k_loop_sum<<<ED / TB, TB, 0, stream>>>(ea, edst, ea + (size_t)N_EDGES * DMODEL);
  k_loop_div<<<ND / TB, TB, 0, stream>>>(ea + (size_t)N_EDGES * DMODEL, cnt);

  // ---- one-time conversions / weight swizzles ----
  k_cvt_f16<<<EAD / TB, TB, 0, stream>>>(ea, ea16, EAD);
  const int WD = NLAYER * DMODEL * DMODEL;
  k_cvt_swizzle_w<<<(WD + TB - 1) / TB, TB, 0, stream>>>(Wl, Wlsw, NLAYER);
  k_cvt_swizzle_w<<<(WD + TB - 1) / TB, TB, 0, stream>>>(Wr, Wrsw, NLAYER);
  k_cvt_swizzle_w<<<(WD + TB - 1) / TB, TB, 0, stream>>>(We, Wesw, NLAYER);

  // ---- layer stack ----
  for (int l = 0; l < NLAYER; ++l) {
    const size_t wl = (size_t)l * DMODEL * DMODEL;
    k_cvt_f16<<<ND / TB, TB, 0, stream>>>(h, h16, ND);

    dim3 gN(N_NODES / 16, 4);
    dim3 gE(EA_EDGES / 16, 4);
    k_gemm_wmma<<<gN, 128, 0, stream>>>(h16, Wlsw + wl, bl + l * DMODEL, xl);
    k_gemm_wmma<<<gN, 128, 0, stream>>>(h16, Wrsw + wl, br + l * DMODEL, xr);
    k_gemm_wmma<<<gE, 128, 0, stream>>>(ea16, Wesw + wl, (const float*)nullptr, ee);

    k_zero_u32<<<(NH + TB - 1) / TB, TB, 0, stream>>>(smaxu, NH);
    k_zero_f32<<<(NH + TB - 1) / TB, TB, 0, stream>>>(denom, NH);
    k_init_out<<<ND / TB, TB, 0, stream>>>(outb, cbias + l * DMODEL);

    k_score_max<<<EAH / TB, TB, 0, stream>>>(xl, xr, ee, att + l * NHEAD * CHEAD,
                                             esrc, edst, score, smaxu);
    k_exp_sum<<<EAH / TB, TB, 0, stream>>>(score, smaxu, denom, edst);
    k_aggregate<<<EAH / TB, TB, 0, stream>>>(score, denom, xl, esrc, edst, outb);

    if (l < NLAYER - 1)
      k_ln_silu<<<N_NODES / 8, 256, 0, stream>>>(outb, ln_g + l * DMODEL,
                                                 ln_b + l * DMODEL, h);
  }

  const int total = ND + N_NODES;
  k_finalize<<<(total + TB - 1) / TB, TB, 0, stream>>>(outb, batch, (float*)d_out);
}
